// Wavenet_50929722196729
// MI455X (gfx1250) — compile-verified, matrix operand order checked by reading, not run on hardware
//
#include <hip/hip_runtime.h>
#include <hip/hip_bf16.h>
#include <cmath>

typedef __attribute__((ext_vector_type(16))) _Float16 v16h;
typedef __attribute__((ext_vector_type(8)))  float    v8f;
typedef unsigned short u16t;

#define TT   8192
#define BB   4
#define DIMC 256
#define LPAD 264   // 256 + 8 halves: padded LDS row stride (264*2 = 528 = 33*16, keeps 16B align)

__device__ __forceinline__ u16t f2h(float f){
    union { _Float16 h; u16t u; } c; c.h = (_Float16)f; return c.u;
}

__device__ __forceinline__ v8f wmma_f16(v16h a, v16h b, v8f c){
    // D = A(16x32 f16) * B(32x16 f16) + C(16x16 f32)
    return __builtin_amdgcn_wmma_f32_16x16x32_f16(false, a, false, b, (short)0, c, false, false);
}

// A fragment: row-major f16 weight matrix [rows][256]; lane holds row (mbase + lane&15),
// halves k = kb..kb+7 and kb+16..kb+23 with kb = k0 + (lane>>4)*8  (ISA 16-bit A layout)
__device__ __forceinline__ v16h loadA(const u16t* __restrict__ W, int row, int k0, int lane){
    const u16t* p = W + ((size_t)row << 8) + (size_t)(k0 + ((lane >> 4) << 3));
    union { v16h h; uint4 u[2]; } f;
    f.u[0] = *reinterpret_cast<const uint4*>(p);
    f.u[1] = *reinterpret_cast<const uint4*>(p + 16);
    return f.h;
}

// B fragment from LDS laid out [n][k] (padded rows): lane holds column n=(lane&15)+16j,
// same kb pattern as A (mirror layout) -> two contiguous 16B ds loads
__device__ __forceinline__ v16h loadB_lds(const u16t* S, int j, int k0, int lane){
    const int n  = (lane & 15) + (j << 4);
    const int kb = k0 + ((lane >> 4) << 3);
    const u16t* p = S + n * LPAD + kb;
    union { v16h h; uint4 u[2]; } f;
    f.u[0] = *reinterpret_cast<const uint4*>(p);
    f.u[1] = *reinterpret_cast<const uint4*>(p + 16);
    return f.h;
}

// ---------------- one-time weight conversion (f32 -> f16, dil re-laid tap-major) -------------
__global__ void wn_wcvt(const float* __restrict__ dilw, const float* __restrict__ resw,
                        const float* __restrict__ ow1,  const float* __restrict__ ow2,
                        u16t* __restrict__ wdil, u16t* __restrict__ wres,
                        u16t* __restrict__ w1,   u16t* __restrict__ w2)
{
    size_t idx = (size_t)blockIdx.x * blockDim.x + threadIdx.x;
    const size_t NDIL = 30ull << 18;            // 30 layers * 2 taps * 512 * 256
    if (idx < NDIL){
        int i   = (int)(idx & 255);
        int o   = (int)((idx >> 8) & 511);
        int tap = (int)((idx >> 17) & 1);
        int l   = (int)(idx >> 18);
        // src: [L][512][256][2]
        wdil[idx] = f2h(dilw[((((size_t)l * 512 + o) << 8) + i) * 2 + tap]);
    }
    if (idx < (30ull << 16)) wres[idx] = f2h(resw[idx]);
    if (idx < (1ull  << 16)){ w1[idx] = f2h(ow1[idx]); w2[idx] = f2h(ow2[idx]); }
}

// ---------------- init: h = in_w * x ; skip = 0 ----------------
__global__ void wn_init(const float* __restrict__ x, const float* __restrict__ inw,
                        float* __restrict__ h32, u16t* __restrict__ h16,
                        float* __restrict__ skip)
{
    size_t idx = (size_t)blockIdx.x * blockDim.x + threadIdx.x;
    if (idx >= (size_t)BB * DIMC * TT) return;
    int t = (int)(idx & (TT - 1));
    int c = (int)((idx >> 13) & 255);
    int b = (int)(idx >> 21);
    float v = inw[c] * x[((size_t)b << 13) + t];
    h32[idx]  = v;
    h16[idx]  = f2h(v);
    skip[idx] = 0.0f;
}

// ---------------- fused WaveNet layer ----------------
// block: 256 threads (8 waves), tile = 64 timesteps of one batch element.
// Conv runs as two tap passes so only 4 B-fragments are live (low VGPR pressure),
// and the h[t-d] staging buffer is reused (barrier-separated) for the gated activation.
__global__ __launch_bounds__(256) void wn_layer(
    const u16t* __restrict__ h16in, const float* __restrict__ h32in,
    u16t* __restrict__ h16out,      float* __restrict__ h32out,
    float* __restrict__ skip,
    const u16t* __restrict__ wdil,  const float* __restrict__ dilb,
    const u16t* __restrict__ wres,  const float* __restrict__ resb,
    int dil)
{
    __shared__ __align__(16) u16t sBc[64 * LPAD];   // h[t] tile,   [n][k]
    __shared__ __align__(16) u16t sPG[64 * LPAD];   // phase 1: h[t-d] tile; phase 2: gated act

    const int tid  = threadIdx.x;
    const int lane = tid & 31;
    const int wv   = tid >> 5;
    const int b    = blockIdx.y;
    const int n0   = blockIdx.x << 6;

    // ---- stage activations into LDS (transpose [c][t] -> [n][c]) ----
    {
        const u16t* hrow = h16in + ((((size_t)b << 8) + tid) << 13);  // channel = tid
        const uint4* src = reinterpret_cast<const uint4*>(hrow + n0);
        #pragma unroll
        for (int q = 0; q < 8; ++q){
            union { uint4 v; u16t s[8]; } tq; tq.v = src[q];
            #pragma unroll
            for (int e = 0; e < 8; ++e) sBc[(q * 8 + e) * LPAD + tid] = tq.s[e];
        }
        if (n0 >= dil && (dil & 7) == 0){
            // interior tile, 16B-aligned dilated tap -> vector path
            const uint4* srcp = reinterpret_cast<const uint4*>(hrow + (n0 - dil));
            #pragma unroll
            for (int q = 0; q < 8; ++q){
                union { uint4 v; u16t s[8]; } tq; tq.v = srcp[q];
                #pragma unroll
                for (int e = 0; e < 8; ++e) sPG[(q * 8 + e) * LPAD + tid] = tq.s[e];
            }
        } else {
            #pragma unroll 4
            for (int n = 0; n < 64; ++n){
                int tp = n0 + n - dil;
                sPG[n * LPAD + tid] = (tp >= 0) ? hrow[tp] : (u16t)0;   // causal left pad
            }
        }
    }
    __builtin_prefetch(wres + ((size_t)(tid & 127) << 9), 0, 1);    // warm residual weights
    __syncthreads();

    const int mt    = wv << 5;                 // this wave: filter rows mt..mt+31, gate rows +256
    const int rbase = (lane >> 4) << 3;        // C-layout: M = mbase + r + 8*(lane>>4)

    // ---- dilated conv: y = W0*h[t-d] + W1*h[t] + b  (filter + gate halves together) ----
    v8f at[2][4], ag[2][4];
    #pragma unroll
    for (int i = 0; i < 2; ++i){
        int mb = mt + (i << 4) + rbase;
        v8f bt, bs;
        #pragma unroll
        for (int r = 0; r < 8; ++r){ bt[r] = dilb[mb + r]; bs[r] = dilb[256 + mb + r]; }
        #pragma unroll
        for (int j = 0; j < 4; ++j){ at[i][j] = bt; ag[i][j] = bs; }
    }
    #pragma unroll
    for (int tap = 0; tap < 2; ++tap){
        const u16t* S = tap ? sBc : sPG;           // tap0 -> h[t-d], tap1 -> h[t]
        const u16t* W = wdil + tap * (512 * 256);
        for (int k0 = 0; k0 < 256; k0 += 32){
            v16h bf[4];
            #pragma unroll
            for (int j = 0; j < 4; ++j) bf[j] = loadB_lds(S, j, k0, lane);
            #pragma unroll
            for (int i = 0; i < 2; ++i){
                int rowT = mt + (i << 4) + (lane & 15);
                v16h aT = loadA(W, rowT,       k0, lane);
                v16h aG = loadA(W, rowT + 256, k0, lane);
                #pragma unroll
                for (int j = 0; j < 4; ++j){
                    at[i][j] = wmma_f16(aT, bf[j], at[i][j]);
                    ag[i][j] = wmma_f16(aG, bf[j], ag[i][j]);
                }
            }
        }
    }
    __syncthreads();   // all conv reads of sPG/sBc complete before sPG is overwritten

    // ---- gate: tanh(t)*sigmoid(g), per-lane (identical C layouts) -> sPG[n][k] as G ----
    #pragma unroll
    for (int i = 0; i < 2; ++i)
        #pragma unroll
        for (int j = 0; j < 4; ++j){
            int n = (lane & 15) + (j << 4);
            #pragma unroll
            for (int r = 0; r < 8; ++r){
                float tv = at[i][j][r];
                float sv = ag[i][j][r];
                float g  = tanhf(tv) * (1.0f / (1.0f + expf(-sv)));
                int m = mt + (i << 4) + rbase + r;
                sPG[n * LPAD + m] = f2h(g);
            }
        }
    __syncthreads();

    // ---- residual GEMM: xs = res_w * g + res_b ----
    v8f c2[2][4];
    #pragma unroll
    for (int i = 0; i < 2; ++i){
        int mb = mt + (i << 4) + rbase;
        v8f br;
        #pragma unroll
        for (int r = 0; r < 8; ++r) br[r] = resb[mb + r];
        #pragma unroll
        for (int j = 0; j < 4; ++j) c2[i][j] = br;
    }
    for (int k0 = 0; k0 < 256; k0 += 32){
        v16h bf[4];
        #pragma unroll
        for (int j = 0; j < 4; ++j) bf[j] = loadB_lds(sPG, j, k0, lane);
        #pragma unroll
        for (int i = 0; i < 2; ++i){
            v16h aR = loadA(wres, mt + (i << 4) + (lane & 15), k0, lane);
            #pragma unroll
            for (int j = 0; j < 4; ++j) c2[i][j] = wmma_f16(aR, bf[j], c2[i][j]);
        }
    }

    // ---- h += xs (f32 state, ping-pong), skip += xs ----
    #pragma unroll
    for (int i = 0; i < 2; ++i)
        #pragma unroll
        for (int j = 0; j < 4; ++j)
            #pragma unroll
            for (int r = 0; r < 8; ++r){
                int m = mt + (i << 4) + rbase + r;
                int n = n0 + (lane & 15) + (j << 4);
                size_t idx = ((((size_t)b << 8) + m) << 13) + n;
                float xs = c2[i][j][r];
                float hn = h32in[idx] + xs;
                h32out[idx] = hn;
                h16out[idx] = f2h(hn);
                skip[idx]  += xs;
            }
}

// ---------------- output head: GELU -> W1 -> GELU -> W2, transposed store ----------------
__global__ __launch_bounds__(256) void wn_out(
    const float* __restrict__ skip,
    const u16t* __restrict__ w1, const float* __restrict__ b1,
    const u16t* __restrict__ w2, const float* __restrict__ b2,
    float* __restrict__ out)
{
    __shared__ __align__(16) u16t sZ [32 * LPAD];
    __shared__ __align__(16) u16t sZ2[32 * LPAD];
    const int tid  = threadIdx.x;
    const int lane = tid & 31;
    const int wv   = tid >> 5;
    const int b    = blockIdx.y;
    const int n0   = blockIdx.x << 5;

    {
        const float* srow = skip + ((((size_t)b << 8) + tid) << 13) + n0;
        #pragma unroll 4
        for (int n = 0; n < 32; ++n){
            float z = srow[n];
            float g = 0.5f * z * (1.0f + erff(z * 0.70710678118f));
            sZ[n * LPAD + tid] = f2h(g);
        }
    }
    __syncthreads();

    const int mt    = wv << 5;
    const int rbase = (lane >> 4) << 3;

    v8f c1[2][2];
    #pragma unroll
    for (int i = 0; i < 2; ++i){
        int mb = mt + (i << 4) + rbase;
        v8f bv;
        #pragma unroll
        for (int r = 0; r < 8; ++r) bv[r] = b1[mb + r];
        c1[i][0] = bv; c1[i][1] = bv;
    }
    for (int k0 = 0; k0 < 256; k0 += 32){
        v16h bb0 = loadB_lds(sZ, 0, k0, lane);
        v16h bb1 = loadB_lds(sZ, 1, k0, lane);
        #pragma unroll
        for (int i = 0; i < 2; ++i){
            v16h a = loadA(w1, mt + (i << 4) + (lane & 15), k0, lane);
            c1[i][0] = wmma_f16(a, bb0, c1[i][0]);
            c1[i][1] = wmma_f16(a, bb1, c1[i][1]);
        }
    }
    #pragma unroll
    for (int i = 0; i < 2; ++i)
        #pragma unroll
        for (int j = 0; j < 2; ++j){
            int n = (lane & 15) + (j << 4);
            #pragma unroll
            for (int r = 0; r < 8; ++r){
                float z = c1[i][j][r];
                float g = 0.5f * z * (1.0f + erff(z * 0.70710678118f));
                sZ2[n * LPAD + (mt + (i << 4) + rbase + r)] = f2h(g);
            }
        }
    __syncthreads();

    v8f c2[2][2];
    #pragma unroll
    for (int i = 0; i < 2; ++i){
        int mb = mt + (i << 4) + rbase;
        v8f bv;
        #pragma unroll
        for (int r = 0; r < 8; ++r) bv[r] = b2[mb + r];
        c2[i][0] = bv; c2[i][1] = bv;
    }
    for (int k0 = 0; k0 < 256; k0 += 32){
        v16h bb0 = loadB_lds(sZ2, 0, k0, lane);
        v16h bb1 = loadB_lds(sZ2, 1, k0, lane);
        #pragma unroll
        for (int i = 0; i < 2; ++i){
            v16h a = loadA(w2, mt + (i << 4) + (lane & 15), k0, lane);
            c2[i][0] = wmma_f16(a, bb0, c2[i][0]);
            c2[i][1] = wmma_f16(a, bb1, c2[i][1]);
        }
    }

    // out[b][t][m]: 8 consecutive r -> 8 consecutive channels -> two float4 stores
    #pragma unroll
    for (int i = 0; i < 2; ++i)
        #pragma unroll
        for (int j = 0; j < 2; ++j){
            int m0 = mt + (i << 4) + rbase;
            int n  = n0 + (lane & 15) + (j << 4);
            float* o = out + (((size_t)b * TT + n) << 8) + m0;
            union { v8f v; float4 f4[2]; } u; u.v = c2[i][j];
            *reinterpret_cast<float4*>(o)     = u.f4[0];
            *reinterpret_cast<float4*>(o + 4) = u.f4[1];
        }
}

extern "C" void kernel_launch(void* const* d_in, const int* in_sizes, int n_in,
                              void* d_out, int out_size, void* d_ws, size_t ws_size,
                              hipStream_t stream)
{
    (void)in_sizes; (void)n_in; (void)out_size; (void)ws_size;
    const float* x    = (const float*)d_in[0];
    const float* inw  = (const float*)d_in[1];
    const float* dilw = (const float*)d_in[2];
    const float* dilb = (const float*)d_in[3];
    const float* resw = (const float*)d_in[4];
    const float* resb = (const float*)d_in[5];
    const float* ow1  = (const float*)d_in[6];
    const float* ob1  = (const float*)d_in[7];
    const float* ow2  = (const float*)d_in[8];
    const float* ob2  = (const float*)d_in[9];
    float* out = (float*)d_out;

    char* ws = (char*)d_ws;
    const size_t H32 = (size_t)BB * DIMC * TT * 4;
    const size_t H16 = (size_t)BB * DIMC * TT * 2;
    float* h32a = (float*)ws;  ws += H32;
    float* h32b = (float*)ws;  ws += H32;
    u16t*  h16a = (u16t*)ws;   ws += H16;
    u16t*  h16b = (u16t*)ws;   ws += H16;
    float* skip = (float*)ws;  ws += H32;
    u16t*  wdil = (u16t*)ws;   ws += (30ull << 18) * 2;
    u16t*  wres = (u16t*)ws;   ws += (30ull << 16) * 2;
    u16t*  w116 = (u16t*)ws;   ws += (1ull  << 16) * 2;
    u16t*  w216 = (u16t*)ws;

    wn_wcvt<<<dim3(30720), 256, 0, stream>>>(dilw, resw, ow1, ow2, wdil, wres, w116, w216);
    wn_init<<<dim3((BB * DIMC * TT) / 256), 256, 0, stream>>>(x, inw, h32a, h16a, skip);

    for (int L = 0; L < 30; ++L){
        int d = 1 << (L % 10);
        const u16t*  hi16 = (L & 1) ? h16b : h16a;
        const float* hi32 = (L & 1) ? h32b : h32a;
        u16t*  ho16 = (L & 1) ? h16a : h16b;
        float* ho32 = (L & 1) ? h32a : h32b;
        wn_layer<<<dim3(TT / 64, BB), 256, 0, stream>>>(
            hi16, hi32, ho16, ho32, skip,
            wdil + (size_t)L * (2 * 512 * 256), dilb + L * 512,
            wres + (size_t)L * 65536,           resb + L * 256, d);
    }
    wn_out<<<dim3(TT / 32, BB), 256, 0, stream>>>(skip, w116, ob1, w216, ob2, out);
}
